// BlineAttention_87840671137864
// MI455X (gfx1250) — matrix operand order repaired and checked by reading, not compiled, not run on hardware
//
#include <hip/hip_runtime.h>

// ---------------------------------------------------------------------------
// MI455X (gfx1250, wave32) implementation of the gated row-attention block.
// All GEMMs run on v_wmma_f32_16x16x32_bf16 (fp32 accumulate); softmax, bias,
// depthwise conv and epilogues in fp32 VALU. Head dim = 32 == one WMMA K-step.
// ---------------------------------------------------------------------------

typedef __attribute__((ext_vector_type(16))) __bf16 v16bf;
typedef __attribute__((ext_vector_type(8)))  float  v8f;

union FragAB { v16bf v; unsigned u[8]; };
union FragC  { v8f   v; float    f[8]; };

__device__ __forceinline__ void fzero(FragC& c) {
#pragma unroll
  for (int i = 0; i < 8; ++i) c.f[i] = 0.0f;
}

__device__ __forceinline__ v8f wmma_bf16(const FragAB& a, const FragAB& b, v8f c) {
  return __builtin_amdgcn_wmma_f32_16x16x32_bf16(false, a.v, false, b.v,
                                                 (short)0, c, false, false);
}

constexpr int B_     = 2;
constexpr int C_     = 192;
constexpr int H_     = 128;
constexpr int W_     = 256;
constexpr int HEADS_ = 6;
constexpr int HD_    = 32;          // head dim
constexpr int HW_    = H_ * W_;     // 32768 pixels per channel plane

// ---------------------------------------------------------------------------
// f32 -> bf16 conversion
// ---------------------------------------------------------------------------
__global__ void f32_to_bf16_kernel(const float* __restrict__ src,
                                   __bf16* __restrict__ dst, size_t n) {
  size_t i = (size_t)blockIdx.x * blockDim.x + threadIdx.x;
  if (i < n) dst[i] = (__bf16)src[i];
}

// ---------------------------------------------------------------------------
// conv1x1 as GEMM: Out[b, m, p] = sum_k W[m,k] * X[b,k,p]   (K = 192 fixed)
// Block tile: M=64, N=128 pixels, K staged 32 at a time in LDS; K-loop fully
// unrolled (6 steps x 4 WMMA). 256 threads = 8 waves; wave w owns n-subtile w,
// all 4 m-subtiles. Next B-tile prefetched via global_prefetch_b8.
// MODE: 0 = bf16 out, 1 = silu->bf16, 2 = (*gate)->bf16, 3 = f32 out
// ---------------------------------------------------------------------------
template <int MODE>
__global__ __launch_bounds__(256) void conv1x1_wmma_kernel(
    const __bf16* __restrict__ X, const __bf16* __restrict__ Wm,
    void* __restrict__ Out, const __bf16* __restrict__ Gate, int Cout) {
  constexpr int Cin = C_;
  __shared__ __align__(16) __bf16 As[64 * 34];    // [m][k], pad 2
  __shared__ __align__(16) __bf16 Bs[128 * 34];   // [pixel][k] (transposed), pad 2

  const int b  = blockIdx.z;
  const int p0 = blockIdx.x * 128;
  const int m0 = blockIdx.y * 64;
  const int t    = threadIdx.x;
  const int wave = t >> 5;
  const int lane = t & 31;
  const int half = lane >> 4;
  const int lr   = lane & 15;

  const __bf16* Xb = X + (size_t)b * Cin * HW_;

  FragC acc[4];
#pragma unroll
  for (int ms = 0; ms < 4; ++ms) fzero(acc[ms]);

#pragma unroll
  for (int ks = 0; ks < Cin; ks += 32) {
    __syncthreads();
    // stage A: W[m0+m][ks..ks+31]  (64x32)
    {
      int m  = t >> 2;
      int k0 = (t & 3) * 8;
      const __bf16* src = Wm + (size_t)(m0 + m) * Cin + ks + k0;
#pragma unroll
      for (int e = 0; e < 8; ++e) As[m * 34 + k0 + e] = src[e];
    }
    // stage B transposed: Bs[pl][k] = X[ks+k][p0+pl]  (32x128)
    {
      int k   = t >> 3;
      int pl0 = (t & 7) * 16;
      const __bf16* src = Xb + (size_t)(ks + k) * HW_ + p0 + pl0;
#pragma unroll
      for (int e = 0; e < 16; ++e) Bs[(pl0 + e) * 34 + k] = src[e];
      // prefetch next K-slab of B into cache (global_prefetch_b8)
      if (ks + 32 < Cin)
        __builtin_prefetch(src + (size_t)32 * HW_, 0, 3);
    }
    __syncthreads();

    // B fragment for this wave's n-subtile (K pairs contiguous in Bs rows)
    FragAB fb;
#pragma unroll
    for (int v = 0; v < 8; ++v) {
      int kk = half * 16 + 2 * v;
      fb.u[v] = *(const unsigned*)&Bs[(wave * 16 + lr) * 34 + kk];
    }
#pragma unroll
    for (int ms = 0; ms < 4; ++ms) {
      FragAB fa;
#pragma unroll
      for (int v = 0; v < 8; ++v) {
        int k0 = half * 16 + 2 * v;
        fa.u[v] = *(const unsigned*)&As[(ms * 16 + lr) * 34 + k0];
      }
      acc[ms].v = wmma_bf16(fa, fb, acc[ms].v);
    }
  }

  // epilogue: D layout -> lane element (M = r + 8*half, N = lane&15)
#pragma unroll
  for (int ms = 0; ms < 4; ++ms) {
#pragma unroll
    for (int r = 0; r < 8; ++r) {
      int    M    = m0 + ms * 16 + half * 8 + r;
      size_t N    = (size_t)p0 + wave * 16 + lr;
      size_t oidx = ((size_t)b * Cout + M) * HW_ + N;
      float  val  = acc[ms].f[r];
      if constexpr (MODE == 0) {
        ((__bf16*)Out)[oidx] = (__bf16)val;
      } else if constexpr (MODE == 1) {             // SiLU
        float s = val / (1.0f + __expf(-val));
        ((__bf16*)Out)[oidx] = (__bf16)s;
      } else if constexpr (MODE == 2) {             // * gate
        float g = (float)Gate[oidx];
        ((__bf16*)Out)[oidx] = (__bf16)(val * g);
      } else {                                      // f32 out
        ((float*)Out)[oidx] = val;
      }
    }
  }
}

// ---------------------------------------------------------------------------
// depthwise 3x3, SAME zero padding, groups = 3C; scales q channels (<C) by
// hd^-0.5 (reference scales q after the depthwise conv).
// ---------------------------------------------------------------------------
__global__ __launch_bounds__(256) void dw3x3_kernel(
    const __bf16* __restrict__ tin, const float* __restrict__ wd,
    __bf16* __restrict__ outp, size_t n) {
  size_t idx = (size_t)blockIdx.x * blockDim.x + threadIdx.x;
  if (idx >= n) return;
  int    x  = (int)(idx % W_);
  size_t r  = idx / W_;
  int    y  = (int)(r % H_);
  size_t bc = r / H_;                 // b*3C + ch
  int    ch = (int)(bc % (3 * C_));
  const __bf16* plane = tin + bc * (size_t)HW_;
  const float*  wk    = wd + (size_t)ch * 9;
  float acc = 0.0f;
#pragma unroll
  for (int dy = 0; dy < 3; ++dy) {
    int yy = y + dy - 1;
    if (yy < 0 || yy >= H_) continue;
#pragma unroll
    for (int dx = 0; dx < 3; ++dx) {
      int xx = x + dx - 1;
      if (xx < 0 || xx >= W_) continue;
      acc += wk[dy * 3 + dx] * (float)plane[(size_t)yy * W_ + xx];
    }
  }
  if (ch < C_) acc *= 0.17677669529663687f;   // 32^-0.5 applied to q
  outp[idx] = (__bf16)acc;
}

// ---------------------------------------------------------------------------
// Row attention: 1 block per (b, y, head). 128 threads = 4 waves; each wave
// handles 4 row-tiles of 16 queries. S = Q K^T (+bias) -> softmax -> O = P V.
// K staged in LDS; Q and V fragments read straight from global (V's K-pairs
// are contiguous along j); P round-trips through per-wave LDS to convert
// D-layout -> A-layout. All WMMA loops fully unrolled.
// ---------------------------------------------------------------------------
__global__ __launch_bounds__(128) void attn_row_kernel(
    const __bf16* __restrict__ qkv, const float* __restrict__ rpb,
    __bf16* __restrict__ outp) {
  __shared__ __align__(16) __bf16 Ks[256 * 34];       // [j][d], pad 2
  __shared__ __align__(16) __bf16 Pb[4 * 16 * 258];   // per-wave [m][j], pad 2

  int bid  = blockIdx.x;
  int head = bid % HEADS_;
  int tmp  = bid / HEADS_;
  int y    = tmp % H_;
  int b    = tmp / H_;

  const int t    = threadIdx.x;
  const int wave = t >> 5;
  const int lane = t & 31;
  const int half = lane >> 4;
  const int lr   = lane & 15;

  size_t base = (size_t)b * (3 * C_) * HW_ + (size_t)y * W_;
  const __bf16* Qg = qkv + base + (size_t)(head * HD_) * HW_;
  const __bf16* Kg = qkv + base + (size_t)(C_ + head * HD_) * HW_;
  const __bf16* Vg = qkv + base + (size_t)(2 * C_ + head * HD_) * HW_;

  // stage K: Ks[j][d]
  {
    int d  = t & 31;
    int j0 = (t >> 5) * 64;
    const __bf16* src = Kg + (size_t)d * HW_ + j0;
    for (int e = 0; e < 64; ++e) Ks[(j0 + e) * 34 + d] = src[e];
  }
  __syncthreads();

  __bf16* Pw = &Pb[wave * 16 * 258];

  for (int rt = 0; rt < 4; ++rt) {
    int i0 = (wave * 4 + rt) * 16;

    // Q fragment: rows i0..i0+15, K = head dim (strided in global, cached)
    FragAB fq;
#pragma unroll
    for (int v = 0; v < 8; ++v) {
      int k0 = half * 16 + 2 * v;
      fq.v[2 * v]     = Qg[(size_t)k0 * HW_ + i0 + lr];
      fq.v[2 * v + 1] = Qg[(size_t)(k0 + 1) * HW_ + i0 + lr];
    }

    // S row-block: 16 tiles of 16x16, one WMMA each (K = 32 = head dim)
    FragC s[16];
#pragma unroll
    for (int nt = 0; nt < 16; ++nt) {
      FragAB fk;
#pragma unroll
      for (int v = 0; v < 8; ++v) {
        int kk = half * 16 + 2 * v;
        fk.u[v] = *(const unsigned*)&Ks[(nt * 16 + lr) * 34 + kk];
      }
      FragC z; fzero(z);
      s[nt].v = wmma_bf16(fq, fk, z.v);
    }

    // bias + running row max
    float rmax[8], rsum[8];
#pragma unroll
    for (int r = 0; r < 8; ++r) rmax[r] = -1e30f;
#pragma unroll
    for (int nt = 0; nt < 16; ++nt) {
#pragma unroll
      for (int r = 0; r < 8; ++r) {
        int   i   = i0 + half * 8 + r;
        int   j   = nt * 16 + lr;
        float val = s[nt].f[r] + rpb[(i - j + (W_ - 1)) * HEADS_ + head];
        s[nt].f[r] = val;
        rmax[r]    = fmaxf(rmax[r], val);
      }
    }
#pragma unroll
    for (int r = 0; r < 8; ++r)
      for (int m = 1; m < 16; m <<= 1)
        rmax[r] = fmaxf(rmax[r], __shfl_xor(rmax[r], m, 32));

#pragma unroll
    for (int r = 0; r < 8; ++r) rsum[r] = 0.0f;
#pragma unroll
    for (int nt = 0; nt < 16; ++nt) {
#pragma unroll
      for (int r = 0; r < 8; ++r) {
        float e = __expf(s[nt].f[r] - rmax[r]);
        s[nt].f[r] = e;
        rsum[r] += e;
      }
    }
#pragma unroll
    for (int r = 0; r < 8; ++r)
      for (int m = 1; m < 16; m <<= 1)
        rsum[r] += __shfl_xor(rsum[r], m, 32);

    // spill P (bf16) to per-wave LDS: Pw[m][j]
#pragma unroll
    for (int nt = 0; nt < 16; ++nt) {
#pragma unroll
      for (int r = 0; r < 8; ++r) {
        int ml = half * 8 + r;
        Pw[ml * 258 + nt * 16 + lr] = (__bf16)s[nt].f[r];
      }
    }
    __syncthreads();

    // O = P V : 2 output tiles (head dim 32), 8 K-steps of 32 over j
    FragC o[2];
    fzero(o[0]); fzero(o[1]);
#pragma unroll
    for (int ksj = 0; ksj < 8; ++ksj) {
      FragAB fp;
#pragma unroll
      for (int v = 0; v < 8; ++v) {
        int k0 = half * 16 + 2 * v;
        fp.u[v] = *(const unsigned*)&Pw[lr * 258 + ksj * 32 + k0];
      }
#pragma unroll
      for (int nt2 = 0; nt2 < 2; ++nt2) {
        FragAB fv;
        int d = nt2 * 16 + lr;
#pragma unroll
        for (int v = 0; v < 8; ++v) {
          int kj = ksj * 32 + half * 16 + 2 * v;   // even -> 4B aligned
          fv.u[v] = *(const unsigned*)&Vg[(size_t)d * HW_ + kj];
        }
        o[nt2].v = wmma_bf16(fp, fv, o[nt2].v);
      }
    }

    // normalize and store: out[b, head*32+d, y, i]
#pragma unroll
    for (int nt2 = 0; nt2 < 2; ++nt2) {
#pragma unroll
      for (int r = 0; r < 8; ++r) {
        int   i   = i0 + half * 8 + r;
        int   d   = nt2 * 16 + lr;
        float val = o[nt2].f[r] / rsum[r];
        outp[((size_t)b * C_ + head * HD_ + d) * HW_ + (size_t)y * W_ + i] =
            (__bf16)val;
      }
    }
    __syncthreads();
  }
}

// ---------------------------------------------------------------------------
// host-side orchestration
// ---------------------------------------------------------------------------
extern "C" void kernel_launch(void* const* d_in, const int* in_sizes, int n_in,
                              void* d_out, int out_size, void* d_ws, size_t ws_size,
                              hipStream_t stream) {
  const float* x     = (const float*)d_in[0];
  const float* rpb   = (const float*)d_in[1];
  const float* wqkv  = (const float*)d_in[2];
  const float* wdep  = (const float*)d_in[3];
  const float* wpre  = (const float*)d_in[4];
  const float* wout  = (const float*)d_in[5];
  const float* wgate = (const float*)d_in[6];
  (void)in_sizes; (void)n_in; (void)out_size; (void)ws_size;

  char*  ws  = (char*)d_ws;
  size_t off = 0;
  auto take = [&](size_t elems) -> __bf16* {
    __bf16* p = (__bf16*)(ws + off);
    off += (elems * sizeof(__bf16) + 255) & ~(size_t)255;
    return p;
  };
  const size_t nX = (size_t)B_ * C_ * HW_;        // 12.58M
  const size_t n3 = (size_t)B_ * 3 * C_ * HW_;    // 37.75M
  __bf16* xb     = take(nX);
  __bf16* tb     = take(n3);
  __bf16* qkvb   = take(n3);
  __bf16* gateb  = take(nX);
  __bf16* attb   = take(nX);
  __bf16* pgb    = take(nX);
  __bf16* wqkvb  = take((size_t)3 * C_ * C_);
  __bf16* wpreb  = take((size_t)C_ * C_);
  __bf16* woutb  = take((size_t)C_ * C_);
  __bf16* wgateb = take((size_t)C_ * C_);

  auto cvt = [&](const float* s, __bf16* d, size_t n) {
    f32_to_bf16_kernel<<<(unsigned)((n + 255) / 256), 256, 0, stream>>>(s, d, n);
  };
  cvt(x, xb, nX);
  cvt(wqkv, wqkvb, (size_t)3 * C_ * C_);
  cvt(wpre, wpreb, (size_t)C_ * C_);
  cvt(wout, woutb, (size_t)C_ * C_);
  cvt(wgate, wgateb, (size_t)C_ * C_);

  dim3 blk(256);
  dim3 gq(HW_ / 128, (3 * C_) / 64, B_);
  conv1x1_wmma_kernel<0><<<gq, blk, 0, stream>>>(xb, wqkvb, tb, nullptr, 3 * C_);

  dw3x3_kernel<<<(unsigned)(n3 / 256), 256, 0, stream>>>(tb, wdep, qkvb, n3);

  dim3 gg(HW_ / 128, C_ / 64, B_);
  conv1x1_wmma_kernel<1><<<gg, blk, 0, stream>>>(xb, wgateb, gateb, nullptr, C_);

  attn_row_kernel<<<B_ * H_ * HEADS_, 128, 0, stream>>>(qkvb, rpb, attb);

  conv1x1_wmma_kernel<2><<<gg, blk, 0, stream>>>(attb, wpreb, pgb, gateb, C_);
  conv1x1_wmma_kernel<3><<<gg, blk, 0, stream>>>(pgb, woutb, d_out, nullptr, C_);
}
